// ScaledDotProductAttention_54374285967459
// MI455X (gfx1250) — matrix operand order repaired and checked by reading, not compiled, not run on hardware
//
#include <hip/hip_runtime.h>

typedef __attribute__((ext_vector_type(16))) _Float16 v16h;
typedef __attribute__((ext_vector_type(8)))  _Float16 v8h;
typedef __attribute__((ext_vector_type(2)))  _Float16 v2h;
typedef __attribute__((ext_vector_type(2)))  __fp16   v2fp16;
typedef __attribute__((ext_vector_type(8)))  float    v8f;

#define N_HEADS 16
#define QLEN    2048
#define SLEN    2048
#define DHEAD   64
// scale folded into Q at load: SCALE * log2(e) so softmax runs in exp2 domain
#define SCALE_LOG2E (0.125f * 1.44269504088896340736f)

#define Q_TILE  128         // rows per block (8 waves x 16 rows)
#define S_TILE  64          // keys per iteration (K-dim of PV split in 2x32)
#define NWAVES  8

#define NEGINF (-__builtin_inff())

#if __has_builtin(__builtin_amdgcn_exp2f)
#define EXP2(x) __builtin_amdgcn_exp2f(x)
#else
#define EXP2(x) exp2f(x)
#endif

union V16 { v16h v; v8h h[2]; };

__device__ __forceinline__ v8h load_cvt8_scaled(const float* p, float s) {
    const float4 f0 = *reinterpret_cast<const float4*>(p);
    const float4 f1 = *reinterpret_cast<const float4*>(p + 4);
    v8h r;
    r[0] = (_Float16)(f0.x * s); r[1] = (_Float16)(f0.y * s);
    r[2] = (_Float16)(f0.z * s); r[3] = (_Float16)(f0.w * s);
    r[4] = (_Float16)(f1.x * s); r[5] = (_Float16)(f1.y * s);
    r[6] = (_Float16)(f1.z * s); r[7] = (_Float16)(f1.w * s);
    return r;
}

__device__ __forceinline__ v2h pack_f16(float a, float b) {
#if __has_builtin(__builtin_amdgcn_cvt_pkrtz)
    const v2fp16 t = __builtin_amdgcn_cvt_pkrtz(a, b);   // v_cvt_pk_rtz_f16_f32
    v2h r;
    __builtin_memcpy(&r, &t, sizeof(r));
    return r;
#else
    v2h r; r[0] = (_Float16)a; r[1] = (_Float16)b; return r;
#endif
}

// key permutation inside a 64-key tile (interleaved logical K)
__device__ __forceinline__ int key_perm(int key) {
    const int t = key >> 4, r = key & 15;
    return (t >> 1) * 32 + 2 * r + (t & 1);
}

// next active S-tile (block-uniform): prefix tiles, plus the 2 diagonal tiles
__device__ __forceinline__ int next_active(int s, int plen, int qbase) {
    if (s < plen)           return s;
    if (s <= qbase)         return qbase;           // qbase is 64-aligned
    if (s <= qbase + 64)    return qbase + 64;
    return SLEN;
}

// one lane copies 16 bytes global -> LDS via the async engine (ASYNCcnt)
__device__ __forceinline__ void async_copy16(unsigned lds_off, const void* g) {
    asm volatile("global_load_async_to_lds_b128 %0, %1, off"
                 :: "v"(lds_off), "v"((unsigned long long)(uintptr_t)g)
                 : "memory");
}
__device__ __forceinline__ void wait_async0() {
    asm volatile("s_wait_asynccnt 0x0" ::: "memory");
}

// ---------------- pre-pass: f32 -> f16, V tile-transposed + key-permuted ------
__global__ __launch_bounds__(256)
void cvt_kv_kernel(const float* __restrict__ Kg, const float* __restrict__ Vg,
                   void* __restrict__ ws) {
    _Float16* Kh = reinterpret_cast<_Float16*>(ws);
    _Float16* Vh = Kh + (size_t)N_HEADS * SLEN * DHEAD;

    const int tid = threadIdx.x;
#pragma unroll
    for (int i = 0; i < 4; ++i) {
        const size_t p = (size_t)blockIdx.x * 1024 + tid + i * 256; // float2 pairs
        const size_t g = p >> 5;            // global row index: n*SLEN + s
        const int    d = (int)(p & 31) * 2;

        const float2 kk = *reinterpret_cast<const float2*>(Kg + g * DHEAD + d);
        *reinterpret_cast<v2h*>(&Kh[g * DHEAD + d]) = pack_f16(kk.x, kk.y);

        const float2 vv = *reinterpret_cast<const float2*>(Vg + g * DHEAD + d);
        const int    key = (int)(g & 63);
        const int    kp  = key_perm(key);
        const size_t vtb = (g & ~(size_t)63) * DHEAD;   // tile base in Vh
        Vh[vtb + (size_t)d * S_TILE + kp]       = (_Float16)vv.x;
        Vh[vtb + (size_t)(d + 1) * S_TILE + kp] = (_Float16)vv.y;
    }
}

// ---------------- main flash-attention kernel ---------------------------------
__global__ __launch_bounds__(256)
void fa_block_diag_kernel(const float* __restrict__ Qg,
                          const float* __restrict__ Kg,
                          const float* __restrict__ Vg,
                          const int*   __restrict__ plen_n,
                          float*       __restrict__ Og,
                          const void*  __restrict__ ws,
                          int preconv) {
    __shared__ __align__(16) _Float16 Klds[2][S_TILE * DHEAD];     // 2x8 KB [key][d]
    __shared__ __align__(16) _Float16 Vlds[2][DHEAD * S_TILE];     // 2x8 KB [d][perm(key)]
    __shared__ __align__(16) _Float16 Plds[NWAVES * 16 * S_TILE];  // 16 KB per-wave P

    const _Float16* Kh = reinterpret_cast<const _Float16*>(ws);
    const _Float16* Vh = Kh + (size_t)N_HEADS * SLEN * DHEAD;

    const int tid   = threadIdx.x;
    const int lane  = tid & 31;
    const int wave  = tid >> 5;
    const int hi    = (lane >> 4) & 1;
    const int lan16 = lane & 15;

    const int qtiles = QLEN / Q_TILE;
    const int n      = blockIdx.x / qtiles;
    const int qbase  = (blockIdx.x % qtiles) * Q_TILE;
    const int qrow0  = qbase + wave * 16;
    const int plen   = plen_n[n];

    // ---- Q A-fragments (pre-scaled by SCALE*log2e) ----
    const size_t qoff = ((size_t)n * QLEN + (qrow0 + lan16)) * DHEAD;
    V16 qa[2];
#pragma unroll
    for (int f = 0; f < 2; ++f) {
        qa[f].h[0] = load_cvt8_scaled(Qg + qoff + f * 32 + hi * 8, SCALE_LOG2E);
        qa[f].h[1] = load_cvt8_scaled(Qg + qoff + f * 32 + 16 + hi * 8, SCALE_LOG2E);
    }

    v16h ones;
#pragma unroll
    for (int e = 0; e < 16; ++e) ones[e] = (_Float16)1.0f;

    float mrun[8], lrun[8];
    v8f   oacc[4];
#pragma unroll
    for (int j = 0; j < 8; ++j) { mrun[j] = NEGINF; lrun[j] = 0.0f; }
#pragma unroll
    for (int nd = 0; nd < 4; ++nd)
        oacc[nd] = (v8f){0.f, 0.f, 0.f, 0.f, 0.f, 0.f, 0.f, 0.f};

    const size_t kvbase = (size_t)n * SLEN * DHEAD;
    const int    p_off  = wave * (16 * S_TILE);

    // staging helper (block-uniform s0), writes buffer b
    auto stage = [&](int s0, int b) {
        if (preconv) {
            const char* ks = (const char*)(Kh + kvbase + (size_t)s0 * DHEAD);
            const char* vs = (const char*)(Vh + kvbase + (size_t)s0 * DHEAD);
            const unsigned kd = (unsigned)(uintptr_t)(&Klds[b][0]) + tid * 16;
            const unsigned vd = (unsigned)(uintptr_t)(&Vlds[b][0]) + tid * 16;
            async_copy16(kd,        ks + tid * 16);
            async_copy16(kd + 4096, ks + tid * 16 + 4096);
            async_copy16(vd,        vs + tid * 16);
            async_copy16(vd + 4096, vs + tid * 16 + 4096);
        } else {
#pragma unroll
            for (int i = 0; i < 8; ++i) {
                const int p   = tid + i * 256;
                const int key = p >> 5;
                const int d   = (p & 31) * 2;
                const int kp  = key_perm(key);
                const float2 kk = *reinterpret_cast<const float2*>(
                    Kg + kvbase + (size_t)(s0 + key) * DHEAD + d);
                *reinterpret_cast<v2h*>(&Klds[b][key * DHEAD + d]) = pack_f16(kk.x, kk.y);
                const float2 vv = *reinterpret_cast<const float2*>(
                    Vg + kvbase + (size_t)(s0 + key) * DHEAD + d);
                Vlds[b][d * S_TILE + kp]       = (_Float16)vv.x;
                Vlds[b][(d + 1) * S_TILE + kp] = (_Float16)vv.y;
            }
        }
    };

    // ---- software pipeline over active tiles, one barrier per tile ----
    int s_cur = next_active(0, plen, qbase);   // diagonal guarantees s_cur < SLEN
    int buf   = 0;
    stage(s_cur, buf);

    while (s_cur < SLEN) {
        const int s_nxt = next_active(s_cur + S_TILE, plen, qbase);

        if (preconv) wait_async0();   // my async stores to LDS done
        __syncthreads();              // everyone's staging done; prev compute done

        if (s_nxt < SLEN) stage(s_nxt, buf ^ 1);   // overlap with compute below

        // ---- scores (exp2 domain): 4 C tiles, K-dim 64 split in 2 ----
        v8f sc[4];
#pragma unroll
        for (int t = 0; t < 4; ++t) {
            v8f acc = (v8f){0.f, 0.f, 0.f, 0.f, 0.f, 0.f, 0.f, 0.f};
#pragma unroll
            for (int c = 0; c < 2; ++c) {
                V16 kb;
                const int krow = (t * 16 + lan16) * DHEAD + c * 32 + hi * 8;
                kb.h[0] = *reinterpret_cast<const v8h*>(&Klds[buf][krow]);
                kb.h[1] = *reinterpret_cast<const v8h*>(&Klds[buf][krow + 16]);
                acc = __builtin_amdgcn_wmma_f32_16x16x32_f16(
                          false, qa[c].v, false, kb.v, (short)0, acc, false, false);
            }
            sc[t] = acc;
        }

        // ---- mask only on boundary tiles (block-uniform) ----
        if (!(s_cur + S_TILE <= plen)) {
#pragma unroll
            for (int t = 0; t < 4; ++t) {
                const int  s_idx  = s_cur + t * 16 + lan16;
                const bool kvalid = (s_idx < plen);
#pragma unroll
                for (int j = 0; j < 8; ++j) {
                    const bool ok = kvalid || (s_idx == qrow0 + j + 8 * hi);
                    sc[t][j] = ok ? sc[t][j] : NEGINF;
                }
            }
        }

        // ---- row max: elementwise across tiles, packed-f16 half-wave reduce ----
        float tmx[8];
#pragma unroll
        for (int j = 0; j < 8; ++j)
            tmx[j] = fmaxf(fmaxf(sc[0][j], sc[1][j]), fmaxf(sc[2][j], sc[3][j]));

        v2h pk[4];
#pragma unroll
        for (int i = 0; i < 4; ++i) pk[i] = pack_f16(tmx[2 * i], tmx[2 * i + 1]);
#pragma unroll
        for (int m = 1; m <= 8; m <<= 1) {
#pragma unroll
            for (int i = 0; i < 4; ++i) {
                int w;  __builtin_memcpy(&w, &pk[i], 4);
                int o = __shfl_xor(w, m, 32);
                v2h q; __builtin_memcpy(&q, &o, 4);
                pk[i][0] = (pk[i][0] >= q[0]) ? pk[i][0] : q[0];
                pk[i][1] = (pk[i][1] >= q[1]) ? pk[i][1] : q[1];
            }
        }
#pragma unroll
        for (int j = 0; j < 8; ++j) tmx[j] = (float)pk[j >> 1][j & 1];

        float alpha[8], safe[8];
#pragma unroll
        for (int j = 0; j < 8; ++j) {
            const float mnew = fmaxf(mrun[j], tmx[j]);
            const float sm   = (mnew == NEGINF) ? 0.0f : mnew;
            alpha[j] = (mrun[j] == NEGINF) ? 0.0f : EXP2(mrun[j] - sm);
            mrun[j]  = mnew;
            safe[j]  = sm;
        }
#pragma unroll
        for (int t = 0; t < 4; ++t)
#pragma unroll
            for (int j = 0; j < 8; ++j)
                sc[t][j] = EXP2(sc[t][j] - safe[j]);   // exp2(-inf) = 0 handles mask

        // ---- P -> LDS (A layout, interleaved logical K) ----
#pragma unroll
        for (int j = 0; j < 8; ++j) {
            const int row = p_off + (j + 8 * hi) * S_TILE;
            *reinterpret_cast<v2h*>(&Plds[row + 2 * lan16])      = pack_f16(sc[0][j], sc[1][j]);
            *reinterpret_cast<v2h*>(&Plds[row + 32 + 2 * lan16]) = pack_f16(sc[2][j], sc[3][j]);
        }

        V16 pa[2];
#pragma unroll
        for (int c = 0; c < 2; ++c) {
            const int prow = p_off + lan16 * S_TILE + c * 32;
            pa[c].h[0] = *reinterpret_cast<const v8h*>(&Plds[prow + hi * 8]);
            pa[c].h[1] = *reinterpret_cast<const v8h*>(&Plds[prow + 16 + hi * 8]);
        }

        // ---- row sums via WMMA against ones ----
        v8f psum = (v8f){0.f, 0.f, 0.f, 0.f, 0.f, 0.f, 0.f, 0.f};
        psum = __builtin_amdgcn_wmma_f32_16x16x32_f16(false, pa[0].v, false, ones,
                                                      (short)0, psum, false, false);
        psum = __builtin_amdgcn_wmma_f32_16x16x32_f16(false, pa[1].v, false, ones,
                                                      (short)0, psum, false, false);
#pragma unroll
        for (int j = 0; j < 8; ++j) lrun[j] = lrun[j] * alpha[j] + psum[j];
#pragma unroll
        for (int nd = 0; nd < 4; ++nd)
#pragma unroll
            for (int j = 0; j < 8; ++j)
                oacc[nd][j] *= alpha[j];

        // ---- O += P * V ----
#pragma unroll
        for (int nd = 0; nd < 4; ++nd) {
#pragma unroll
            for (int c = 0; c < 2; ++c) {
                V16 vb;
                const int vrow = (nd * 16 + lan16) * S_TILE + c * 32 + hi * 8;
                vb.h[0] = *reinterpret_cast<const v8h*>(&Vlds[buf][vrow]);
                vb.h[1] = *reinterpret_cast<const v8h*>(&Vlds[buf][vrow + 16]);
                oacc[nd] = __builtin_amdgcn_wmma_f32_16x16x32_f16(
                               false, pa[c].v, false, vb.v, (short)0, oacc[nd], false, false);
            }
        }

        buf ^= 1;
        s_cur = s_nxt;
    }

    // ---- epilogue ----
#pragma unroll
    for (int j = 0; j < 8; ++j) {
        const float inv = 1.0f / lrun[j];               // diagonal guarantees lrun > 0
        const int   qm  = qrow0 + j + 8 * hi;
        const size_t ob = ((size_t)n * QLEN + qm) * DHEAD;
#pragma unroll
        for (int nd = 0; nd < 4; ++nd)
            Og[ob + nd * 16 + lan16] = oacc[nd][j] * inv;
    }
}

extern "C" void kernel_launch(void* const* d_in, const int* in_sizes, int n_in,
                              void* d_out, int out_size, void* d_ws, size_t ws_size,
                              hipStream_t stream) {
    (void)in_sizes; (void)n_in; (void)out_size;
    const float* Qg = (const float*)d_in[0];
    const float* Kg = (const float*)d_in[1];
    const float* Vg = (const float*)d_in[2];
    const int*   pl = (const int*)d_in[3];
    float*       Og = (float*)d_out;

    const size_t need = (size_t)2 * N_HEADS * SLEN * DHEAD * 2; // Kh + Vh, f16
    const int preconv = (d_ws != nullptr && ws_size >= need) ? 1 : 0;

    if (preconv) {
        dim3 cgrid((N_HEADS * SLEN * DHEAD / 2) / 1024);   // 1024 blocks
        cvt_kv_kernel<<<cgrid, 256, 0, stream>>>(Kg, Vg, d_ws);
    }

    dim3 grid(N_HEADS * (QLEN / Q_TILE));   // 256 workgroups
    dim3 block(256);
    fa_block_diag_kernel<<<grid, block, 0, stream>>>(Qg, Kg, Vg, pl, Og, d_ws, preconv);
}
